// ProjBlock4_58025008169500
// MI455X (gfx1250) — matrix operand order repaired and checked by reading, not compiled, not run on hardware
//
#include <hip/hip_runtime.h>
#include <hip/hip_bf16.h>

typedef __attribute__((ext_vector_type(16))) _Float16 v16h;
typedef __attribute__((ext_vector_type(8)))  float    v8f;
typedef __attribute__((ext_vector_type(2)))  _Float16 h2;
typedef unsigned int u32;

// Pointer types matching the async-load builtin's expected signature:
// param0: int4* in global (AS1), param1: int4* in LDS (AS3).
typedef int v4i __attribute__((vector_size(16)));
typedef __attribute__((address_space(1))) v4i* gptr4;
typedef __attribute__((address_space(3))) v4i* lptr4;

#if __has_builtin(__builtin_amdgcn_global_load_async_to_lds_b128)
#define HAVE_ASYNC_LDS 1
#else
#define HAVE_ASYNC_LDS 0
#endif

// 16-byte global -> LDS copy. Async DMA path (ASYNCcnt) when available.
__device__ __forceinline__ void cp16(_Float16* ldst, const _Float16* gsrc) {
#if HAVE_ASYNC_LDS
  __builtin_amdgcn_global_load_async_to_lds_b128(
      (gptr4)(unsigned long long)gsrc,
      (lptr4)(unsigned)(unsigned long long)ldst, 0, 0);
#else
  *(uint4*)ldst = *(const uint4*)gsrc;
#endif
}

__device__ __forceinline__ void cp_wait() {
#if HAVE_ASYNC_LDS
#if __has_builtin(__builtin_amdgcn_s_wait_asynccnt)
  __builtin_amdgcn_s_wait_asynccnt(0);
#else
  asm volatile("s_wait_asynccnt 0x0" ::: "memory");
#endif
#endif
}

// ---------------------------------------------------------------------------
// WMMA helpers (CDNA5 wave32: D(16x16 f32) = A(16x32 f16) x B(32x16 f16) + C)
// ---------------------------------------------------------------------------
__device__ __forceinline__ v8f wmma_f16(v16h a, v16h b, v8f c) {
  return __builtin_amdgcn_wmma_f32_16x16x32_f16(
      /*neg_a=*/false, a, /*neg_b=*/false, b,
      /*c_mod=*/(short)0, c, /*reuse_a=*/false, /*reuse_b=*/false);
}

// A fragment: 16(M) x 32(K) f16. lane: m = lane&15, kh = lane>>4.
// VGPR v holds K = (v>>2)*16 + kh*8 + (v&3)*2 and K+1 (ISA A layout).
__device__ __forceinline__ v16h frag_a(const _Float16* base, int stride, int lane) {
  const int m = lane & 15, kh = lane >> 4;
  v16h a;
#pragma unroll
  for (int v = 0; v < 8; ++v) {
    const int k = ((v >> 2) << 4) + (kh << 3) + ((v & 3) << 1);
    const h2 p = *(const h2*)(base + m * stride + k);
    a[2 * v] = p.x;
    a[2 * v + 1] = p.y;
  }
  return a;
}

// B fragment: 32(K) x 16(N) f16. lane: n = lane&15, kh = lane>>4.
// VGPR j holds K = (j>>2)*16 + kh*8 + (j&3)*2 and K+1 for column n (LDS stored [n][k]).
__device__ __forceinline__ v16h frag_b(const _Float16* base, int stride, int lane) {
  const int n = lane & 15, kh = lane >> 4;
  v16h b;
#pragma unroll
  for (int j = 0; j < 8; ++j) {
    const int k = ((j >> 2) << 4) + (kh << 3) + ((j & 3) << 1);
    const h2 p = *(const h2*)(base + n * stride + k);
    b[2 * j] = p.x;
    b[2 * j + 1] = p.y;
  }
  return b;
}

// C/D 16x16 f32 layout: lane: n = lane&15; VGPR r <-> M = r + 8*(lane>>4).

// ---------------------------------------------------------------------------
// Kernel 1: q/k/v projections.  out(128 x 4096) = W(128x128) * x(128x4096) + b
// grid = (32 token tiles of 128, 3 projections, 4 batches), block = 256 (8 waves)
// Q/K written as [b*2+h][l][d] f16, V as [b*2+h][d][l] f16.
// ---------------------------------------------------------------------------
__global__ __launch_bounds__(256) void proj_kernel(
    const float* __restrict__ x,
    const float* __restrict__ wq, const float* __restrict__ bq,
    const float* __restrict__ wk, const float* __restrict__ bk,
    const float* __restrict__ wv, const float* __restrict__ bv,
    _Float16* __restrict__ Qws, _Float16* __restrict__ Kws,
    _Float16* __restrict__ Vws) {
  __shared__ __align__(16) _Float16 Wt[128 * 32];  // [c_out][k]  (A side)
  __shared__ __align__(16) _Float16 Xt[128 * 32];  // [token][k]  (B side)
  const int t = threadIdx.x, lane = t & 31, wave = t >> 5;
  const int ntile = blockIdx.x;  // token block of 128
  const int p = blockIdx.y;      // 0=q 1=k 2=v
  const int b = blockIdx.z;
  const int l0 = ntile * 128;
  const float* W = (p == 0) ? wq : (p == 1) ? wk : wv;
  const float* Bi = (p == 0) ? bq : (p == 1) ? bk : bv;
  const float* xb = x + (size_t)b * 128 * 4096;

  v8f acc[8];
#pragma unroll
  for (int nt = 0; nt < 8; ++nt)
#pragma unroll
    for (int r = 0; r < 8; ++r) acc[nt][r] = 0.f;

  for (int kb = 0; kb < 128; kb += 32) {
    __syncthreads();
#pragma unroll
    for (int i = 0; i < 8; ++i) {  // 2048 f32 pairs of W -> packed f16 b32
      const int idx = t + 256 * i;
      const int k2 = idx & 15, oo = idx >> 4;
      const float2 w2 = *(const float2*)(W + oo * 128 + kb + k2 * 2);
      h2 pk;
      pk.x = (_Float16)w2.x;
      pk.y = (_Float16)w2.y;
      *(h2*)(Wt + oo * 32 + k2 * 2) = pk;
    }
#pragma unroll
    for (int i = 0; i < 8; ++i) {  // x transposed: pair along k (two rows)
      const int idx = t + 256 * i;
      const int tok = idx & 127, k2 = idx >> 7;  // 0..15
      const float x0 = xb[(size_t)(kb + 2 * k2) * 4096 + l0 + tok];
      const float x1 = xb[(size_t)(kb + 2 * k2 + 1) * 4096 + l0 + tok];
      h2 pk;
      pk.x = (_Float16)x0;
      pk.y = (_Float16)x1;
      *(h2*)(Xt + tok * 32 + 2 * k2) = pk;
    }
    __syncthreads();
    const v16h a = frag_a(Wt + (wave * 16) * 32, 32, lane);
    v16h bf[8];
#pragma unroll
    for (int nt = 0; nt < 8; ++nt) bf[nt] = frag_b(Xt + (nt * 16) * 32, 32, lane);
#pragma unroll
    for (int nt = 0; nt < 8; ++nt) acc[nt] = wmma_f16(a, bf[nt], acc[nt]);
  }

  const int mh = lane >> 4, nl = lane & 15;
#pragma unroll
  for (int nt = 0; nt < 8; ++nt) {
    const int ltok = l0 + nt * 16 + nl;
#pragma unroll
    for (int r = 0; r < 8; ++r) {
      const int c = wave * 16 + r + 8 * mh;  // output channel
      const float v = acc[nt][r] + Bi[c];
      const int hh = c >> 6, dd = c & 63;
      const size_t bh = (size_t)(b * 2 + hh);
      const _Float16 hv = (_Float16)v;
      if (p == 0)
        Qws[(bh * 4096 + ltok) * 64 + dd] = hv;
      else if (p == 1)
        Kws[(bh * 4096 + ltok) * 64 + dd] = hv;
      else
        Vws[(bh * 64 + dd) * 4096 + ltok] = hv;
    }
  }
}

// ---------------------------------------------------------------------------
// Kernel 2: flash attention per (b,h).  grid = (32 query blocks of 128, 8 bh)
// block = 256 (8 waves, 16 queries per wave). Streams 128-key blocks.
// Computes S^T = K * Q^T so the S^T C-tiles ARE the A-fragments of P for the
// P*V^T WMMA (no LDS transpose). Output stored as (B, C=128, H=32, W=128) f16.
// ---------------------------------------------------------------------------
__global__ __launch_bounds__(256) void attn_kernel(
    const _Float16* __restrict__ Qws, const _Float16* __restrict__ Kws,
    const _Float16* __restrict__ Vws, _Float16* __restrict__ Ows) {
  __shared__ __align__(16) _Float16 Kt[128 * 64];  // [key][d]; also Q staging
  __shared__ __align__(16) _Float16 Vt[64 * 128];  // [d][key]
  const int t = threadIdx.x, lane = t & 31, wave = t >> 5;
  const int kh = lane >> 4, nl = lane & 15;
  const int qb = blockIdx.x * 128;
  const int bh = blockIdx.y;  // b*2 + h
  const _Float16* Qg = Qws + (size_t)bh * 4096 * 64;
  const _Float16* Kg = Kws + (size_t)bh * 4096 * 64;
  const _Float16* Vg = Vws + (size_t)bh * 64 * 4096;

  // stage Q tile [128][64] via Kt, pull per-wave B fragments (d chunks 0/32)
#pragma unroll
  for (int i = 0; i < 4; ++i) {
    const int idx = t + 256 * i;  // 1024 x 16B
    cp16(Kt + idx * 8, Qg + (size_t)qb * 64 + idx * 8);
  }
  cp_wait();
  __syncthreads();
  const v16h bq0 = frag_b(Kt + (wave * 16) * 64, 64, lane);
  const v16h bq1 = frag_b(Kt + (wave * 16) * 64 + 32, 64, lane);
  __syncthreads();

  float mrow = -1e30f, lrow = 0.f;  // softmax state for query q = nl
  v8f o[4];
#pragma unroll
  for (int dt = 0; dt < 4; ++dt)
#pragma unroll
    for (int r = 0; r < 8; ++r) o[dt][r] = 0.f;

  for (int kb = 0; kb < 4096; kb += 128) {
    __syncthreads();
    // K tile [128 keys][64 d], V tile [64 d][128 keys]: global -> LDS
#pragma unroll
    for (int i = 0; i < 4; ++i) {
      const int idx = t + 256 * i;  // 1024 x 16B
      cp16(Kt + idx * 8, Kg + (size_t)kb * 64 + idx * 8);
    }
#pragma unroll
    for (int i = 0; i < 4; ++i) {
      const int idx = t + 256 * i;
      const int mu = idx & 15, dd = idx >> 4;
      cp16(Vt + dd * 128 + mu * 8, Vg + (size_t)dd * 4096 + kb + mu * 8);
    }
    // prefetch next key block into caches (global_prefetch_b8 path)
    if (kb + 128 < 4096) {
      __builtin_prefetch((const char*)(Kg + (size_t)(kb + 128) * 64) + t * 64, 0, 1);
      __builtin_prefetch((const char*)(Vg + (size_t)(t >> 2) * 4096 + (kb + 128) +
                                       (t & 3) * 32), 0, 1);
    }
    cp_wait();
    __syncthreads();

    // S^T = K * Q^T : 8 key tiles (M = key), software-pipelined A fragments
    v8f s[8];
    v16h a0 = frag_a(Kt, 64, lane);
    v16h a1 = frag_a(Kt + 32, 64, lane);
#pragma unroll
    for (int nt = 0; nt < 8; ++nt) {
      v16h p0, p1;
      if (nt < 7) {
        p0 = frag_a(Kt + ((nt + 1) * 16) * 64, 64, lane);
        p1 = frag_a(Kt + ((nt + 1) * 16) * 64 + 32, 64, lane);
      }
#pragma unroll
      for (int r = 0; r < 8; ++r) s[nt][r] = 0.f;
      s[nt] = wmma_f16(a0, bq0, s[nt]);
      s[nt] = wmma_f16(a1, bq1, s[nt]);
      if (nt < 7) {
        a0 = p0;
        a1 = p1;
      }
    }

    // online softmax: row q = nl lives on lanes {nl, nl+16}
    float sm = -1e30f;
#pragma unroll
    for (int nt = 0; nt < 8; ++nt)
#pragma unroll
      for (int r = 0; r < 8; ++r) {
        const float sv = s[nt][r] * 0.125f;  // 1/sqrt(64)
        s[nt][r] = sv;
        sm = fmaxf(sm, sv);
      }
    sm = fmaxf(sm, __shfl_xor(sm, 16, 32));
    const float nm = fmaxf(mrow, sm);
    const float alpha = __expf(mrow - nm);
    mrow = nm;
    float rs = 0.f;
#pragma unroll
    for (int nt = 0; nt < 8; ++nt)
#pragma unroll
      for (int r = 0; r < 8; ++r) {
        const float pv = __expf(s[nt][r] - nm);
        s[nt][r] = pv;
        rs += pv;
      }
    rs += __shfl_xor(rs, 16, 32);
    lrow = lrow * alpha + rs;

    // rescale O (rows of O are on the VGPR axis -> broadcast alpha per row)
    float aB[8];
#pragma unroll
    for (int r = 0; r < 8; ++r) aB[r] = __shfl(alpha, r + 8 * kh, 32);
#pragma unroll
    for (int dt = 0; dt < 4; ++dt)
#pragma unroll
      for (int r = 0; r < 8; ++r) o[dt][r] *= aB[r];

    // O^T += P * V^T : P A-fragments come straight from the S^T C-tiles;
    // V B-fragments software-pipelined one step ahead.
    v16h bv = frag_b(Vt, 128, lane);
#pragma unroll
    for (int kc = 0; kc < 4; ++kc) {
      v16h ap;
#pragma unroll
      for (int i = 0; i < 8; ++i) {
        ap[i] = (_Float16)s[2 * kc][i];
        ap[8 + i] = (_Float16)s[2 * kc + 1][i];
      }
#pragma unroll
      for (int dt = 0; dt < 4; ++dt) {
        v16h nb;
        const int nxt = kc * 4 + dt + 1;
        if (nxt < 16)
          nb = frag_b(Vt + ((nxt & 3) * 16) * 128 + (nxt >> 2) * 32, 128, lane);
        o[dt] = wmma_f16(ap, bv, o[dt]);
        if (nxt < 16) bv = nb;
      }
    }
  }

  // normalize rows and scatter to (B,C,H,W) f16
  const float linv = 1.f / lrow;
  float lB[8];
#pragma unroll
  for (int r = 0; r < 8; ++r) lB[r] = __shfl(linv, r + 8 * kh, 32);
  const int b = bh >> 1, hd = bh & 1;
#pragma unroll
  for (int dt = 0; dt < 4; ++dt) {
#pragma unroll
    for (int r = 0; r < 8; ++r) {
      const int ltok = qb + wave * 16 + r + 8 * kh;
      const int c = hd * 64 + dt * 16 + nl;
      const int hi = ltok >> 7, wi = ltok & 127;
      Ows[(((size_t)(b * 128 + c)) * 32 + hi) * 128 + wi] =
          (_Float16)(o[dt][r] * lB[r]);
    }
  }
}

// ---------------------------------------------------------------------------
// Kernel 3: fc over last dim + BatchNorm + PReLU.
// Y(16384 x 128) = G(16384 x 128) * fc_w^T + fc_b, then BN(c) + PReLU. f32 out.
// grid = 128 row-blocks of 128, block = 256.
// ---------------------------------------------------------------------------
__global__ __launch_bounds__(256) void fc_bn_prelu_kernel(
    const _Float16* __restrict__ Ows, const float* __restrict__ fc_w,
    const float* __restrict__ fc_b, const float* __restrict__ bn_gamma,
    const float* __restrict__ bn_beta, const float* __restrict__ bn_mean,
    const float* __restrict__ bn_var, const float* __restrict__ prelu_a,
    float* __restrict__ out) {
  __shared__ __align__(16) _Float16 At[128 * 32];  // [row][w]
  __shared__ __align__(16) _Float16 Bt[128 * 32];  // [o][w]
  const int t = threadIdx.x, lane = t & 31, wave = t >> 5;
  const int mb = blockIdx.x * 128;

  v8f acc[8];
#pragma unroll
  for (int nt = 0; nt < 8; ++nt)
#pragma unroll
    for (int r = 0; r < 8; ++r) acc[nt][r] = 0.f;

  for (int kb = 0; kb < 128; kb += 32) {
    __syncthreads();
#pragma unroll
    for (int i = 0; i < 2; ++i) {  // At: 512 x 16B (f16 already)
      const int idx = t + 256 * i;
      const int ku = idx & 3, m = idx >> 2;
      cp16(At + m * 32 + ku * 8, Ows + (size_t)(mb + m) * 128 + kb + ku * 8);
    }
#pragma unroll
    for (int i = 0; i < 8; ++i) {  // Bt: 2048 f32 pairs -> packed f16 b32
      const int idx = t + 256 * i;
      const int k2 = idx & 15, oo = idx >> 4;
      const float2 w2 = *(const float2*)(fc_w + oo * 128 + kb + k2 * 2);
      h2 pk;
      pk.x = (_Float16)w2.x;
      pk.y = (_Float16)w2.y;
      *(h2*)(Bt + oo * 32 + k2 * 2) = pk;
    }
    cp_wait();
    __syncthreads();
    const v16h a = frag_a(At + (wave * 16) * 32, 32, lane);
    v16h bf[8];
#pragma unroll
    for (int nt = 0; nt < 8; ++nt) bf[nt] = frag_b(Bt + (nt * 16) * 32, 32, lane);
#pragma unroll
    for (int nt = 0; nt < 8; ++nt) acc[nt] = wmma_f16(a, bf[nt], acc[nt]);
  }

  const float pa = prelu_a[0];
  const int mh = lane >> 4, nl = lane & 15;
#pragma unroll
  for (int nt = 0; nt < 8; ++nt) {
    const int oo = nt * 16 + nl;
    const float fb = fc_b[oo];
#pragma unroll
    for (int r = 0; r < 8; ++r) {
      const int rowg = mb + wave * 16 + r + 8 * mh;
      const int c = (rowg >> 5) & 127;  // row = ((b*128 + c)*32 + h)
      float v = acc[nt][r] + fb;
      v = (v - bn_mean[c]) * rsqrtf(bn_var[c] + 1e-5f) * bn_gamma[c] + bn_beta[c];
      v = v > 0.f ? v : pa * v;
      out[(size_t)rowg * 128 + oo] = v;
    }
  }
}

// ---------------------------------------------------------------------------
extern "C" void kernel_launch(void* const* d_in, const int* in_sizes, int n_in,
                              void* d_out, int out_size, void* d_ws,
                              size_t ws_size, hipStream_t stream) {
  const float* x = (const float*)d_in[0];
  const float* wq = (const float*)d_in[1];
  const float* bq = (const float*)d_in[2];
  const float* wk = (const float*)d_in[3];
  const float* bk = (const float*)d_in[4];
  const float* wv = (const float*)d_in[5];
  const float* bv = (const float*)d_in[6];
  const float* fc_w = (const float*)d_in[7];
  const float* fc_b = (const float*)d_in[8];
  const float* bn_gamma = (const float*)d_in[9];
  const float* bn_beta = (const float*)d_in[10];
  const float* bn_mean = (const float*)d_in[11];
  const float* bn_var = (const float*)d_in[12];
  const float* prelu_a = (const float*)d_in[13];
  float* out = (float*)d_out;

  const size_t seg = (size_t)8 * 4096 * 64;  // halves per Q/K/V/O segment
  _Float16* Qws = (_Float16*)d_ws;
  _Float16* Kws = Qws + seg;
  _Float16* Vws = Kws + seg;
  _Float16* Ows = Vws + seg;

  proj_kernel<<<dim3(32, 3, 4), 256, 0, stream>>>(x, wq, bq, wk, bk, wv, bv,
                                                  Qws, Kws, Vws);
  attn_kernel<<<dim3(32, 8), 256, 0, stream>>>(Qws, Kws, Vws, Ows);
  fc_bn_prelu_kernel<<<128, 256, 0, stream>>>(Ows, fc_w, fc_b, bn_gamma,
                                              bn_beta, bn_mean, bn_var,
                                              prelu_a, out);
}
